// Qwen3MoeSparseMoeBlock_46909632807484
// MI455X (gfx1250) — matrix operand order repaired and checked by reading, not compiled, not run on hardware
//
#include <hip/hip_runtime.h>
#include <hip/hip_bf16.h>

// Problem constants (match reference)
#define T_TOK 2048
#define H_DIM 2048
#define E_NUM 64
#define TOPK  8
#define I_DIM 768
#define PADCAP (T_TOK*TOPK + E_NUM*16)   // 17408 padded token-expert slots
#define MAX_TILES 1152                   // >= sum ceil(count_e/16) worst case (1088)
#define PF_DIST 256                      // prefetch distance in floats (8 k-steps ahead)

typedef __attribute__((ext_vector_type(16))) __bf16 v16bf;
typedef __attribute__((ext_vector_type(8)))  __bf16 v8bf;
typedef __attribute__((ext_vector_type(8)))  float  v8f;

// ---- workspace layout (4-byte word offsets into d_ws) ----
constexpr int WS_TOPK_I = 0;                          // T*K ints
constexpr int WS_TOPK_W = WS_TOPK_I + T_TOK*TOPK;     // T*K floats
constexpr int WS_COUNTS = WS_TOPK_W + T_TOK*TOPK;     // E ints
constexpr int WS_PADOFF = WS_COUNTS + E_NUM;          // E+1 ints (padded offsets)
constexpr int WS_CURSOR = WS_PADOFF + E_NUM + 1;      // E ints
constexpr int WS_NTILES = WS_CURSOR + E_NUM;          // 1 int
constexpr int WS_TILE_E = WS_NTILES + 1;              // MAX_TILES ints
constexpr int WS_TILE_M = WS_TILE_E + MAX_TILES;      // MAX_TILES ints
constexpr int WS_TOKEN  = WS_TILE_M + MAX_TILES;      // PADCAP ints
constexpr int WS_WEIGHT = WS_TOKEN + PADCAP;          // PADCAP floats
constexpr int WS_HDR_END = WS_WEIGHT + PADCAP;
constexpr size_t WS_H_BYTE = (((size_t)WS_HDR_END * 4) + 255) & ~(size_t)255; // bf16 h buffer

// ---- helpers ----
__device__ __forceinline__ v8f wmma_bf16(v16bf a, v16bf b, v8f c) {
  return __builtin_amdgcn_wmma_f32_16x16x32_bf16(false, a, false, b, (short)0, c, false, false);
}

__device__ __forceinline__ v16bf pack16(float4 a0, float4 a1, float4 a2, float4 a3) {
  v16bf v;
  v[0]=(__bf16)a0.x;  v[1]=(__bf16)a0.y;  v[2]=(__bf16)a0.z;  v[3]=(__bf16)a0.w;
  v[4]=(__bf16)a1.x;  v[5]=(__bf16)a1.y;  v[6]=(__bf16)a1.z;  v[7]=(__bf16)a1.w;
  v[8]=(__bf16)a2.x;  v[9]=(__bf16)a2.y;  v[10]=(__bf16)a2.z; v[11]=(__bf16)a2.w;
  v[12]=(__bf16)a3.x; v[13]=(__bf16)a3.y; v[14]=(__bf16)a3.z; v[15]=(__bf16)a3.w;
  return v;
}

// ---- kernels ----
__global__ void k_init(int* __restrict__ wsw) {
  int i = blockIdx.x * blockDim.x + threadIdx.x;
  int stride = gridDim.x * blockDim.x;
  for (int j = i; j < PADCAP; j += stride) {
    wsw[WS_TOKEN + j] = 0;
    ((float*)wsw)[WS_WEIGHT + j] = 0.0f;
  }
  if (i < E_NUM) wsw[WS_COUNTS + i] = 0;
}

__global__ void k_zero_out(float* __restrict__ out) {
  int i = blockIdx.x * blockDim.x + threadIdx.x;
  if (i < T_TOK * H_DIM) out[i] = 0.0f;
}

__global__ __launch_bounds__(64)
void k_router(const float* __restrict__ x, const float* __restrict__ wr, int* __restrict__ wsw) {
  __shared__ float xs[H_DIM];
  __shared__ float lg[E_NUM];
  int t = blockIdx.x;
  const float* xrow = x + (size_t)t * H_DIM;
  for (int i = threadIdx.x; i < H_DIM; i += 64) xs[i] = xrow[i];
  __syncthreads();
  int e = threadIdx.x;   // 64 threads, one expert each
  const float* w = wr + (size_t)e * H_DIM;
  float s = 0.0f;
  for (int i = 0; i < H_DIM; i += 4) {
    s += xs[i]*w[i] + xs[i+1]*w[i+1] + xs[i+2]*w[i+2] + xs[i+3]*w[i+3];
  }
  lg[e] = s;
  __syncthreads();
  if (threadIdx.x == 0) {
    float mx = lg[0];
    for (int j = 1; j < E_NUM; j++) mx = fmaxf(mx, lg[j]);
    float p[E_NUM]; float sum = 0.0f;
    for (int j = 0; j < E_NUM; j++) { p[j] = __expf(lg[j] - mx); sum += p[j]; }
    float inv = 1.0f / sum;
    int idxs[TOPK]; float wts[TOPK]; float tsum = 0.0f;
    for (int k = 0; k < TOPK; k++) {
      int bi = 0; float bv = -1.0f;
      for (int j = 0; j < E_NUM; j++) if (p[j] > bv) { bv = p[j]; bi = j; }
      idxs[k] = bi; wts[k] = bv * inv; tsum += bv * inv; p[bi] = -2.0f;
    }
    float invt = 1.0f / tsum;
    for (int k = 0; k < TOPK; k++) {
      wsw[WS_TOPK_I + t*TOPK + k] = idxs[k];
      ((float*)wsw)[WS_TOPK_W + t*TOPK + k] = wts[k] * invt;
      atomicAdd(&wsw[WS_COUNTS + idxs[k]], 1);
    }
  }
}

__global__ void k_scan(int* __restrict__ wsw) {
  if (threadIdx.x != 0 || blockIdx.x != 0) return;
  int run = 0, nt = 0;
  for (int e = 0; e < E_NUM; e++) {
    wsw[WS_PADOFF + e] = run;
    wsw[WS_CURSOR + e] = run;
    int c = wsw[WS_COUNTS + e];
    int tiles = (c + 15) >> 4;
    for (int m = 0; m < tiles; m++) {
      if (nt < MAX_TILES) { wsw[WS_TILE_E + nt] = e; wsw[WS_TILE_M + nt] = m; nt++; }
    }
    run += tiles * 16;
  }
  wsw[WS_PADOFF + E_NUM] = run;
  wsw[WS_NTILES] = nt;
}

__global__ void k_scatter(int* __restrict__ wsw) {
  int i = blockIdx.x * blockDim.x + threadIdx.x;
  if (i >= T_TOK * TOPK) return;
  int e = wsw[WS_TOPK_I + i];
  float w = ((const float*)wsw)[WS_TOPK_W + i];
  int pos = atomicAdd(&wsw[WS_CURSOR + e], 1);
  wsw[WS_TOKEN + pos] = i / TOPK;
  ((float*)wsw)[WS_WEIGHT + pos] = w;
}

// gate/up grouped GEMM: per block = one (expert, 16-token tile) x 32 columns of I.
// 2-stage software pipeline: load raw data for k+32 while converting+WMMA-ing k.
__global__ __launch_bounds__(32, 1)
void k_gateup(const float* __restrict__ x, const float* __restrict__ wg_all,
              const float* __restrict__ wu_all, int* __restrict__ wsw) {
  int tile = blockIdx.x;
  if (tile >= wsw[WS_NTILES]) return;
  int e = wsw[WS_TILE_E + tile];
  int m = wsw[WS_TILE_M + tile];
  int slot_base = wsw[WS_PADOFF + e] + m * 16;
  int nbase = blockIdx.y * 32;
  int lane = threadIdx.x;
  bool hib = lane >= 16;
  int lrow = lane & 15;

  int tokA = wsw[WS_TOKEN + slot_base + lrow];
  const float* xa  = x + (size_t)tokA * H_DIM + (hib ? 16 : 0);
  const float* wg0 = wg_all + (size_t)e * I_DIM * H_DIM + (size_t)(nbase + lrow) * H_DIM + (hib ? 16 : 0);
  const float* wg1 = wg0 + (size_t)16 * H_DIM;
  const float* wu0 = wu_all + (size_t)e * I_DIM * H_DIM + (size_t)(nbase + lrow) * H_DIM + (hib ? 16 : 0);
  const float* wu1 = wu0 + (size_t)16 * H_DIM;

  v8f ag0 = {}, ag1 = {}, au0 = {}, au1 = {};

  float4 ra[2][4];
  float4 rg0[2][4], rg1[2][4], ru0[2][4], ru1[2][4];

  auto LOAD = [&](int b, int k0) {
    const float4* xq  = (const float4*)(xa  + k0);
    const float4* qg0 = (const float4*)(wg0 + k0);
    const float4* qg1 = (const float4*)(wg1 + k0);
    const float4* qu0 = (const float4*)(wu0 + k0);
    const float4* qu1 = (const float4*)(wu1 + k0);
    ra[b][0] = xq[0]; ra[b][1] = xq[1]; ra[b][2] = xq[2]; ra[b][3] = xq[3];
    #pragma unroll
    for (int j = 0; j < 4; j++) {
      rg0[b][j] = qg0[j]; rg1[b][j] = qg1[j];
      ru0[b][j] = qu0[j]; ru1[b][j] = qu1[j];
    }
    __builtin_prefetch(wg0 + k0 + PF_DIST, 0, 0);
    __builtin_prefetch(wg1 + k0 + PF_DIST, 0, 0);
    __builtin_prefetch(wu0 + k0 + PF_DIST, 0, 0);
    __builtin_prefetch(wu1 + k0 + PF_DIST, 0, 0);
  };
  auto COMPUTE = [&](int b) {
    v16bf A   = pack16(ra[b][0],  ra[b][1],  ra[b][2],  ra[b][3]);
    v16bf Bg0 = pack16(rg0[b][0], rg0[b][1], rg0[b][2], rg0[b][3]);
    v16bf Bg1 = pack16(rg1[b][0], rg1[b][1], rg1[b][2], rg1[b][3]);
    v16bf Bu0 = pack16(ru0[b][0], ru0[b][1], ru0[b][2], ru0[b][3]);
    v16bf Bu1 = pack16(ru1[b][0], ru1[b][1], ru1[b][2], ru1[b][3]);
    ag0 = wmma_bf16(A, Bg0, ag0);
    ag1 = wmma_bf16(A, Bg1, ag1);
    au0 = wmma_bf16(A, Bu0, au0);
    au1 = wmma_bf16(A, Bu1, au1);
  };

  LOAD(0, 0);
  #pragma unroll 2
  for (int k0 = 0; k0 < H_DIM - 32; k0 += 32) {
    int cur = (k0 >> 5) & 1;
    LOAD(cur ^ 1, k0 + 32);   // next k-step's loads in flight during compute
    COMPUTE(cur);
  }
  COMPUTE(((H_DIM - 32) >> 5) & 1);

  __bf16* hbuf = (__bf16*)((char*)wsw + WS_H_BYTE);
  #pragma unroll
  for (int r = 0; r < 8; r++) {
    int slot = slot_base + (hib ? 8 + r : r);
    float g0 = ag0[r], u0 = au0[r], g1 = ag1[r], u1 = au1[r];
    float h0 = (g0 / (1.0f + __expf(-g0))) * u0;   // silu(g)*u
    float h1 = (g1 / (1.0f + __expf(-g1))) * u1;
    hbuf[(size_t)slot * I_DIM + nbase + lrow]      = (__bf16)h0;
    hbuf[(size_t)slot * I_DIM + nbase + 16 + lrow] = (__bf16)h1;
  }
}

// down grouped GEMM: A = h (bf16), B = wd^T slices; same 2-stage pipeline;
// weighted atomic combine into out.
__global__ __launch_bounds__(32, 1)
void k_down(const float* __restrict__ wd_all, float* __restrict__ out, int* __restrict__ wsw) {
  int tile = blockIdx.x;
  if (tile >= wsw[WS_NTILES]) return;
  int e = wsw[WS_TILE_E + tile];
  int m = wsw[WS_TILE_M + tile];
  int slot_base = wsw[WS_PADOFF + e] + m * 16;
  int nbase = blockIdx.y * 32;
  int lane = threadIdx.x;
  bool hib = lane >= 16;
  int lrow = lane & 15;

  const __bf16* hbuf = (const __bf16*)((const char*)wsw + WS_H_BYTE);
  const __bf16* ha = hbuf + (size_t)(slot_base + lrow) * I_DIM + (hib ? 16 : 0);
  const float* wd0 = wd_all + (size_t)e * H_DIM * I_DIM + (size_t)(nbase + lrow) * I_DIM + (hib ? 16 : 0);
  const float* wd1 = wd0 + (size_t)16 * I_DIM;

  v8f a0 = {}, a1 = {};

  v8bf rlo[2], rhi[2];
  float4 rd0[2][4], rd1[2][4];

  auto LOAD = [&](int b, int k0) {
    rlo[b] = *(const v8bf*)(ha + k0);
    rhi[b] = *(const v8bf*)(ha + k0 + 8);
    const float4* qd0 = (const float4*)(wd0 + k0);
    const float4* qd1 = (const float4*)(wd1 + k0);
    #pragma unroll
    for (int j = 0; j < 4; j++) { rd0[b][j] = qd0[j]; rd1[b][j] = qd1[j]; }
    __builtin_prefetch(wd0 + k0 + PF_DIST, 0, 0);
    __builtin_prefetch(wd1 + k0 + PF_DIST, 0, 0);
  };
  auto COMPUTE = [&](int b) {
    v16bf A;
    #pragma unroll
    for (int j = 0; j < 8; j++) { A[j] = rlo[b][j]; A[8 + j] = rhi[b][j]; }
    v16bf B0 = pack16(rd0[b][0], rd0[b][1], rd0[b][2], rd0[b][3]);
    v16bf B1 = pack16(rd1[b][0], rd1[b][1], rd1[b][2], rd1[b][3]);
    a0 = wmma_bf16(A, B0, a0);
    a1 = wmma_bf16(A, B1, a1);
  };

  LOAD(0, 0);
  #pragma unroll 2
  for (int k0 = 0; k0 < I_DIM - 32; k0 += 32) {
    int cur = (k0 >> 5) & 1;
    LOAD(cur ^ 1, k0 + 32);
    COMPUTE(cur);
  }
  COMPUTE(((I_DIM - 32) >> 5) & 1);

  #pragma unroll
  for (int r = 0; r < 8; r++) {
    int slot = slot_base + (hib ? 8 + r : r);
    int tok = wsw[WS_TOKEN + slot];
    float w = ((const float*)wsw)[WS_WEIGHT + slot];
    atomicAdd(&out[(size_t)tok * H_DIM + nbase + lrow],      w * a0[r]);
    atomicAdd(&out[(size_t)tok * H_DIM + nbase + 16 + lrow], w * a1[r]);
  }
}

extern "C" void kernel_launch(void* const* d_in, const int* in_sizes, int n_in,
                              void* d_out, int out_size, void* d_ws, size_t ws_size,
                              hipStream_t stream) {
  const float* x  = (const float*)d_in[0];   // [T,H]
  const float* wr = (const float*)d_in[1];   // [E,H]
  const float* wg = (const float*)d_in[2];   // [E,I,H]
  const float* wu = (const float*)d_in[3];   // [E,I,H]
  const float* wd = (const float*)d_in[4];   // [E,H,I]
  float* out = (float*)d_out;                // [T,H]
  int* wsw = (int*)d_ws;

  k_init    <<<dim3(64),                        dim3(256), 0, stream>>>(wsw);
  k_zero_out<<<dim3((T_TOK*H_DIM + 255)/256),   dim3(256), 0, stream>>>(out);
  k_router  <<<dim3(T_TOK),                     dim3(64),  0, stream>>>(x, wr, wsw);
  k_scan    <<<dim3(1),                         dim3(1),   0, stream>>>(wsw);
  k_scatter <<<dim3((T_TOK*TOPK + 255)/256),    dim3(256), 0, stream>>>(wsw);
  k_gateup  <<<dim3(MAX_TILES, I_DIM/32),       dim3(32),  0, stream>>>(x, wg, wu, wsw);
  k_down    <<<dim3(MAX_TILES, H_DIM/32),       dim3(32),  0, stream>>>(wd, out, wsw);
}
